// OptimizedCustomLSTM2_10471130268442
// MI455X (gfx1250) — matrix operand order repaired and checked by reading, not compiled, not run on hardware
//
#include <hip/hip_runtime.h>
#include <hip/hip_bf16.h>

typedef __bf16 bf16_t;
typedef __attribute__((ext_vector_type(16))) __bf16 v16bf;
typedef __attribute__((ext_vector_type(8)))  __bf16 v8bf;
typedef __attribute__((ext_vector_type(8)))  float  v8f;

namespace {
constexpr int kSeq = 1024;
constexpr int kB   = 64;
constexpr int kI   = 768;
constexpr int kH   = 768;
constexpr int kG   = 4 * kH;   // 3072 gate columns
}

__device__ __forceinline__ float sigmoid_f(float x) {
  return 1.0f / (1.0f + __expf(-x));
}

// ---------------- conversion / init kernels ----------------

// W is (4H, I+H) row-major fp32. Split into Wx (4H,I) and Wh (4H,H), bf16.
__global__ void k_convert_w(const float* __restrict__ W,
                            bf16_t* __restrict__ Wx,
                            bf16_t* __restrict__ Wh) {
  const long n = (long)kG * (kI + kH);
  for (long idx = blockIdx.x * (long)blockDim.x + threadIdx.x; idx < n;
       idx += (long)gridDim.x * blockDim.x) {
    const int row = (int)(idx / (kI + kH));
    const int col = (int)(idx % (kI + kH));
    const float v = W[idx];
    if (col < kI) Wx[(long)row * kI + col]        = (bf16_t)v;
    else          Wh[(long)row * kH + (col - kI)] = (bf16_t)v;
  }
}

__global__ void k_convert_x(const float* __restrict__ x, bf16_t* __restrict__ xb) {
  const long n = (long)kSeq * kB * kI;
  for (long idx = blockIdx.x * (long)blockDim.x + threadIdx.x; idx < n;
       idx += (long)gridDim.x * blockDim.x)
    xb[idx] = (bf16_t)x[idx];
}

__global__ void k_init_hc(bf16_t* __restrict__ h0, bf16_t* __restrict__ h1,
                          float* __restrict__ c) {
  const int n = kB * kH;
  for (int idx = blockIdx.x * blockDim.x + threadIdx.x; idx < n;
       idx += gridDim.x * blockDim.x) {
    h0[idx] = (bf16_t)0.0f;
    h1[idx] = (bf16_t)0.0f;
    c[idx]  = 0.0f;
  }
}

// ---------------- input projection GEMM ----------------
// xproj[sb, g] = sum_k xb[sb,k] * Wx[g,k] + bias[g]
// One wave computes a 16(M) x 64(N) strip: A frag reused over 4 WMMA tiles.
__global__ __launch_bounds__(256)
void k_xproj(const bf16_t* __restrict__ Xb, const bf16_t* __restrict__ Wx,
             const float* __restrict__ bias, float* __restrict__ xproj) {
  const int lane = threadIdx.x & 31;
  const int col  = lane & 15;
  const int hsel = lane >> 4;
  const long N_GROUPS = kG / 64;                       // 48
  const long M_TILES  = (long)kSeq * kB / 16;          // 4096
  const long wid = blockIdx.x * 8L + (threadIdx.x >> 5);
  if (wid >= M_TILES * N_GROUPS) return;
  const int mt = (int)(wid / N_GROUPS);
  const int ng = (int)(wid % N_GROUPS);
  const int m0 = mt * 16;
  const int n0 = ng * 64;

  v8f acc[4];
#pragma unroll
  for (int j = 0; j < 4; ++j)
#pragma unroll
    for (int r = 0; r < 8; ++r) acc[j][r] = 0.0f;

  const bf16_t* arow = Xb + (long)(m0 + col) * kI;
#pragma unroll 4
  for (int k0 = 0; k0 < kI; k0 += 32) {
    v8bf lo = *(const v8bf*)(arow + k0 + hsel * 16);
    v8bf hi = *(const v8bf*)(arow + k0 + hsel * 16 + 8);
    v16bf a;
#pragma unroll
    for (int j = 0; j < 8; ++j) { a[j] = lo[j]; a[j + 8] = hi[j]; }
#pragma unroll
    for (int j = 0; j < 4; ++j) {
      const bf16_t* brow = Wx + (long)(n0 + j * 16 + col) * kI + k0 + hsel * 16;
      v16bf bm = *(const v16bf*)brow;
      acc[j] = __builtin_amdgcn_wmma_f32_16x16x32_bf16(
          false, a, false, bm, (short)0, acc[j], false, false);
    }
  }
#pragma unroll
  for (int j = 0; j < 4; ++j) {
    const int gcol = n0 + j * 16 + col;
    const float bv = bias[gcol];
#pragma unroll
    for (int r = 0; r < 8; ++r)
      xproj[(long)(m0 + r + 8 * hsel) * kG + gcol] = acc[j][r] + bv;
  }
}

// ---------------- fused recurrent step ----------------
// Block = 4 waves; wave w computes the 16x16 tile of gate w (columns w*H+n0..)
// for batch rows m0..m0+15. Gates meet in LDS, then the cell update is done
// in-block and h (bf16, double-buffered) + outputs (f32) are written.
__global__ __launch_bounds__(128)
void k_lstm_step(const float* __restrict__ xproj_t,   // (B, 4H) for this t
                 const bf16_t* __restrict__ Wh,        // (4H, H)
                 const bf16_t* __restrict__ h_in,      // (B, H)
                 bf16_t* __restrict__ h_out,           // (B, H)
                 float* __restrict__ c,                // (B, H), in place
                 float* __restrict__ out,              // (B, SEQ, H)
                 float* __restrict__ hT,               // (B, H) final h
                 float* __restrict__ cT,               // (B, H) final c
                 int t) {
  __shared__ float lds[4 * 256];
  const int lane = threadIdx.x & 31;
  const int w    = threadIdx.x >> 5;      // gate index: i,f,g,o
  const int col  = lane & 15;
  const int hsel = lane >> 4;
  const int n0   = blockIdx.x * 16;       // column within H
  const int m0   = blockIdx.y * 16;       // batch row
  const int gcol0 = w * kH + n0;

  // C initialized with x_proj[t] (bias already folded in).
  v8f acc;
#pragma unroll
  for (int r = 0; r < 8; ++r)
    acc[r] = xproj_t[(long)(m0 + r + 8 * hsel) * kG + gcol0 + col];

  const bf16_t* arow = h_in + (long)(m0 + col) * kH;
  const bf16_t* brow = Wh + (long)(gcol0 + col) * kH + hsel * 16;
#pragma unroll 4
  for (int k0 = 0; k0 < kH; k0 += 32) {
    v8bf lo = *(const v8bf*)(arow + k0 + hsel * 16);
    v8bf hi = *(const v8bf*)(arow + k0 + hsel * 16 + 8);
    v16bf a;
#pragma unroll
    for (int j = 0; j < 8; ++j) { a[j] = lo[j]; a[j + 8] = hi[j]; }
    v16bf bm = *(const v16bf*)(brow + k0);
    acc = __builtin_amdgcn_wmma_f32_16x16x32_bf16(
        false, a, false, bm, (short)0, acc, false, false);
  }

  // C/D layout: vgpr r, lane -> row r+8*hsel, col (lane&15).
#pragma unroll
  for (int r = 0; r < 8; ++r)
    lds[w * 256 + (r + 8 * hsel) * 16 + col] = acc[r];
  __syncthreads();

#pragma unroll
  for (int e = threadIdx.x; e < 256; e += 128) {
    const int row = e >> 4;
    const int cc  = e & 15;
    const float iv = sigmoid_f(lds[e]);
    const float fv = sigmoid_f(lds[256 + e]);
    const float gv = tanhf(lds[512 + e]);
    const float ov = sigmoid_f(lds[768 + e]);
    const long ci = (long)(m0 + row) * kH + (n0 + cc);
    const float cold = c[ci];
    const float cnew = fv * cold + iv * gv;
    const float hnew = ov * tanhf(cnew);
    c[ci] = cnew;
    h_out[ci] = (bf16_t)hnew;
    out[(long)(m0 + row) * kSeq * kH + (long)t * kH + (n0 + cc)] = hnew;
    if (t == kSeq - 1) { hT[ci] = hnew; cT[ci] = cnew; }
  }
}

// ---------------- launcher ----------------
extern "C" void kernel_launch(void* const* d_in, const int* in_sizes, int n_in,
                              void* d_out, int out_size, void* d_ws, size_t ws_size,
                              hipStream_t stream) {
  (void)in_sizes; (void)n_in; (void)out_size; (void)ws_size;
  const float* x = (const float*)d_in[0];   // (SEQ, B, I)
  const float* W = (const float*)d_in[1];   // (4H, I+H)
  const float* b = (const float*)d_in[2];   // (4H)

  float* out = (float*)d_out;                       // (B, SEQ, H)
  float* hT  = out + (long)kB * kSeq * kH;          // (B, H)
  float* cT  = hT + (long)kB * kH;                  // (B, H)

  char* ws = (char*)d_ws;
  size_t off = 0;
  bf16_t* xb = (bf16_t*)(ws + off); off += (size_t)kSeq * kB * kI * sizeof(bf16_t);
  bf16_t* Wx = (bf16_t*)(ws + off); off += (size_t)kG * kI * sizeof(bf16_t);
  bf16_t* Wh = (bf16_t*)(ws + off); off += (size_t)kG * kH * sizeof(bf16_t);
  bf16_t* h_ping = (bf16_t*)(ws + off); off += (size_t)kB * kH * sizeof(bf16_t);
  bf16_t* h_pong = (bf16_t*)(ws + off); off += (size_t)kB * kH * sizeof(bf16_t);
  float*  c      = (float*)(ws + off);  off += (size_t)kB * kH * sizeof(float);
  float*  xproj  = (float*)(ws + off);  off += (size_t)kSeq * kB * kG * sizeof(float);

  k_convert_w<<<2048, 256, 0, stream>>>(W, Wx, Wh);
  k_convert_x<<<4096, 256, 0, stream>>>(x, xb);
  k_init_hc<<<192, 256, 0, stream>>>(h_ping, h_pong, c);

  // 4096 M-tiles * 48 N-groups = 196608 wave tasks / 8 waves per block.
  k_xproj<<<24576, 256, 0, stream>>>(xb, Wx, b, xproj);

  for (int t = 0; t < kSeq; ++t) {
    const bf16_t* hin = (t & 1) ? h_pong : h_ping;
    bf16_t*      hout = (t & 1) ? h_ping : h_pong;
    k_lstm_step<<<dim3(kH / 16, kB / 16), 128, 0, stream>>>(
        xproj + (size_t)t * kB * kG, Wh, hin, hout, c, out, hT, cT, t);
  }
}
